// UPCLoss_3298534883961
// MI455X (gfx1250) — matrix operand order repaired and checked by reading, not compiled, not run on hardware
//
#include <hip/hip_runtime.h>
#include <hip/hip_bf16.h>
#include <stdint.h>

// ---------------------------------------------------------------------------
// UPC loss: mean((dist(i,j) * mask(i,j))^2), dist from Gram G = X @ X^T,
// X: (512, 65536) fp32.
//   1) k_convert : fp32 -> bf16 hi + bf16 lo residual (split precision)
//   2) k_pos     : invert ranking permutation
//   3) k_gram    : split-K bf16 WMMA GEMM (hi*hi + hi*lo + lo*hi), double-
//                  buffered GVS-mode global_load_async_to_lds_b128 staging,
//                  partials -> Gpart[16][512][512] (no atomics, deterministic)
//   4) k_diag    : sq[i] = sum_s Gpart[s][i][i]
//   5) k_loss    : per-element loss + block reduction -> partial[1024]
//   6) k_final   : deterministic tree-sum -> out[0]
// ---------------------------------------------------------------------------

typedef __attribute__((ext_vector_type(16))) __bf16 v16bf;
typedef __attribute__((ext_vector_type(8)))  __bf16 v8bf;
typedef __attribute__((ext_vector_type(8)))  float  v8f;

#define NB2     512
#define KDIM    65536
#define NCLS    16
#define BT      128          // output macro-tile (rows == cols)
#define KS      32           // k per WMMA step
#define SPLITK  16
#define KSLICE  (KDIM / SPLITK)   // 4096
#define NSTEP   (KSLICE / KS)     // 128

// ---- CDNA5 async global->LDS copy, GVS mode (SGPR base + u32 VGPR offset) --
__device__ __forceinline__ void async_b128_gvs(uint32_t lds_dst, uint32_t voff,
                                               const void* sbase) {
  asm volatile("global_load_async_to_lds_b128 %0, %1, %2"
               :: "v"(lds_dst), "v"(voff), "s"(sbase) : "memory");
}
__device__ __forceinline__ void wait_async0() {
  asm volatile("s_wait_asynccnt 0" ::: "memory");
}
__device__ __forceinline__ uint32_t lds_off(const void* p) {
  return (uint32_t)(uintptr_t)p;   // flat addr[31:0] == LDS byte offset
}

// ---- 1) fp32 -> bf16 hi/lo split ------------------------------------------
__global__ __launch_bounds__(256) void k_convert(const float* __restrict__ x,
                                                 __bf16* __restrict__ xh,
                                                 __bf16* __restrict__ xl) {
  size_t base = ((size_t)blockIdx.x * 256 + threadIdx.x) * 8;
  const float4* p = (const float4*)(x + base);
  float4 f0 = p[0], f1 = p[1];
  float v[8] = {f0.x, f0.y, f0.z, f0.w, f1.x, f1.y, f1.z, f1.w};
  v8bf h, l;
#pragma unroll
  for (int i = 0; i < 8; ++i) {
    __bf16 hb = (__bf16)v[i];
    h[i] = hb;
    l[i] = (__bf16)(v[i] - (float)hb);
  }
  *(v8bf*)(xh + base) = h;
  *(v8bf*)(xl + base) = l;
}

// ---- 2) pos[i][v] = position of class v in ranking row i ------------------
__global__ void k_pos(const int* __restrict__ ranking, int* __restrict__ pos) {
  int i = blockIdx.x * blockDim.x + threadIdx.x;
  if (i < NB2) {
#pragma unroll
    for (int k = 0; k < NCLS; ++k) pos[i * NCLS + ranking[i * NCLS + k]] = k;
  }
}

// ---- 3) split-K Gram via bf16 WMMA ----------------------------------------
__global__ __launch_bounds__(256) void k_gram(const __bf16* __restrict__ xh,
                                              const __bf16* __restrict__ xl,
                                              float* __restrict__ gpart) {
  __shared__ __bf16 sAh[2][BT * KS];
  __shared__ __bf16 sAl[2][BT * KS];
  __shared__ __bf16 sBh[2][BT * KS];
  __shared__ __bf16 sBl[2][BT * KS];

  const int tile  = blockIdx.x / SPLITK;     // 0..15
  const int slice = blockIdx.x % SPLITK;     // 0..15
  const int bi = tile >> 2, bj = tile & 3;   // 4x4 blocks of 128
  const int k0base = slice * KSLICE;
  const int tid  = threadIdx.x;
  const int lane = tid & 31;
  const int w    = tid >> 5;                 // wave 0..7
  const int wm   = w >> 1;                   // 0..3 (M direction)
  const int wn   = w & 1;                    // 0..1 (N direction)

  // Per-thread copy geometry: 512 b128 chunks per 128x32 tile; each thread
  // moves 2 chunks per array.  VGPR byte offset is loop-invariant (GVS).
  const int cidx0 = tid, cidx1 = tid + 256;
  const int r0 = cidx0 >> 2, c0 = (cidx0 & 3) * 8;
  const int r1 = cidx1 >> 2, c1 = (cidx1 & 3) * 8;
  const uint32_t voff0 = (uint32_t)(((size_t)r0 * KDIM + c0) * sizeof(__bf16));
  const uint32_t voff1 = (uint32_t)(((size_t)r1 * KDIM + c1) * sizeof(__bf16));
  const int l0 = r0 * KS + c0, l1 = r1 * KS + c1;
  const size_t rowA = (size_t)(bi * BT) * KDIM;
  const size_t rowB = (size_t)(bj * BT) * KDIM;

  auto stage = [&](int buf, int k0) {
    const __bf16* bAh = xh + rowA + k0;      // uniform -> SGPR pair
    const __bf16* bAl = xl + rowA + k0;
    const __bf16* bBh = xh + rowB + k0;
    const __bf16* bBl = xl + rowB + k0;
    async_b128_gvs(lds_off(&sAh[buf][l0]), voff0, bAh);
    async_b128_gvs(lds_off(&sAh[buf][l1]), voff1, bAh);
    async_b128_gvs(lds_off(&sAl[buf][l0]), voff0, bAl);
    async_b128_gvs(lds_off(&sAl[buf][l1]), voff1, bAl);
    async_b128_gvs(lds_off(&sBh[buf][l0]), voff0, bBh);
    async_b128_gvs(lds_off(&sBh[buf][l1]), voff1, bBh);
    async_b128_gvs(lds_off(&sBl[buf][l0]), voff0, bBl);
    async_b128_gvs(lds_off(&sBl[buf][l1]), voff1, bBl);
  };

  stage(0, k0base);
  wait_async0();
  __syncthreads();

  v8f acc[2][4];
#pragma unroll
  for (int mi = 0; mi < 2; ++mi)
#pragma unroll
    for (int ni = 0; ni < 4; ++ni)
#pragma unroll
      for (int e = 0; e < 8; ++e) acc[mi][ni][e] = 0.0f;

  // A frag (16x32), this code's k order (B same): lane L holds row M=L%16; elems 0..15 = K
  // koffA+0..15 contiguous, koffA = (L<16)?0:16.
  // B frag (32x16): lane L holds col N=L%16, K = koffB+0..15 contiguous.
  const int m     = lane & 15;
  const int koffA = (lane < 16) ? 0 : 16;
  const int koffB = (lane < 16) ? 0 : 16;
  const int ab0 = (wm * 32 + 0  + m) * KS + koffA;   // M-tile 0 of this wave
  const int ab1 = (wm * 32 + 16 + m) * KS + koffA;   // M-tile 1

  for (int step = 0; step < NSTEP; ++step) {
    const int buf = step & 1;
    if (step + 1 < NSTEP) stage(buf ^ 1, k0base + (step + 1) * KS);

    v16bf ah[2], al[2];
    {
      v8bf x0 = *(const v8bf*)&sAh[buf][ab0];
      v8bf x1 = *(const v8bf*)&sAh[buf][ab0 + 8];
      ah[0] = __builtin_shufflevector(x0, x1, 0, 1, 2, 3, 4, 5, 6, 7,
                                      8, 9, 10, 11, 12, 13, 14, 15);
      v8bf y0 = *(const v8bf*)&sAh[buf][ab1];
      v8bf y1 = *(const v8bf*)&sAh[buf][ab1 + 8];
      ah[1] = __builtin_shufflevector(y0, y1, 0, 1, 2, 3, 4, 5, 6, 7,
                                      8, 9, 10, 11, 12, 13, 14, 15);
      v8bf z0 = *(const v8bf*)&sAl[buf][ab0];
      v8bf z1 = *(const v8bf*)&sAl[buf][ab0 + 8];
      al[0] = __builtin_shufflevector(z0, z1, 0, 1, 2, 3, 4, 5, 6, 7,
                                      8, 9, 10, 11, 12, 13, 14, 15);
      v8bf u0 = *(const v8bf*)&sAl[buf][ab1];
      v8bf u1 = *(const v8bf*)&sAl[buf][ab1 + 8];
      al[1] = __builtin_shufflevector(u0, u1, 0, 1, 2, 3, 4, 5, 6, 7,
                                      8, 9, 10, 11, 12, 13, 14, 15);
    }
#pragma unroll
    for (int ni = 0; ni < 4; ++ni) {
      const int bbase = ((wn * 64 + ni * 16) + m) * KS + koffB;
      v16bf bh = *(const v16bf*)&sBh[buf][bbase];
      v16bf bl = *(const v16bf*)&sBl[buf][bbase];
#pragma unroll
      for (int mi = 0; mi < 2; ++mi) {
        acc[mi][ni] = __builtin_amdgcn_wmma_f32_16x16x32_bf16(
            false, ah[mi], false, bh, (short)0, acc[mi][ni], false, false);
        acc[mi][ni] = __builtin_amdgcn_wmma_f32_16x16x32_bf16(
            false, ah[mi], false, bl, (short)0, acc[mi][ni], false, false);
        acc[mi][ni] = __builtin_amdgcn_wmma_f32_16x16x32_bf16(
            false, al[mi], false, bh, (short)0, acc[mi][ni], false, false);
      }
    }
    wait_async0();
    __syncthreads();
  }

  // C/D layout: VGPR r -> M = r + (lane<16?0:8), N = lane%16
  float* gp = gpart + (size_t)slice * NB2 * NB2;
  const int rbase = bi * BT + wm * 32 + ((lane < 16) ? 0 : 8);
  const int cbase = bj * BT + wn * 64 + m;
#pragma unroll
  for (int mi = 0; mi < 2; ++mi)
#pragma unroll
    for (int ni = 0; ni < 4; ++ni)
#pragma unroll
      for (int r = 0; r < 8; ++r)
        gp[(size_t)(rbase + mi * 16 + r) * NB2 + cbase + ni * 16] =
            acc[mi][ni][r];
}

// ---- 4) row squared norms from Gram diagonal ------------------------------
__global__ void k_diag(const float* __restrict__ gpart, float* __restrict__ sq) {
  int i = blockIdx.x * blockDim.x + threadIdx.x;
  if (i < NB2) {
    float s = 0.f;
    for (int p = 0; p < SPLITK; ++p)
      s += gpart[(size_t)p * NB2 * NB2 + (size_t)i * NB2 + i];
    sq[i] = s;
  }
}

// ---- 5) per-element loss + per-block reduction ----------------------------
__global__ __launch_bounds__(256) void k_loss(const float* __restrict__ gpart,
                                              const float* __restrict__ sq,
                                              const int* __restrict__ pos,
                                              const int* __restrict__ ranking,
                                              const int* __restrict__ choice,
                                              float* __restrict__ partial) {
  int idx = blockIdx.x * 256 + threadIdx.x;   // 0..262143
  int i = idx >> 9, j = idx & 511;
  float g = 0.f;
  for (int p = 0; p < SPLITK; ++p) g += gpart[(size_t)p * NB2 * NB2 + idx];
  float d2 = sq[i] + sq[j] - 2.0f * g;
  float dist = sqrtf(fmaxf(d2, 1e-12f));
  int lab = ranking[j * NCLS];                // labels[j] = ranking[j][0]
  float jr = (float)pos[i * NCLS + lab];      // rank_of_label[i][j]
  float c = (float)choice[i];
  float msk = (jr < c) ? (1.0f - jr / c)
                       : (-(jr - c + 1.0f) / ((float)NCLS - c));
  float v = dist * msk;
  v = v * v;

  __shared__ float red[256];
  red[threadIdx.x] = v;
  __syncthreads();
  for (int s = 128; s > 0; s >>= 1) {
    if (threadIdx.x < s) red[threadIdx.x] += red[threadIdx.x + s];
    __syncthreads();
  }
  if (threadIdx.x == 0) partial[blockIdx.x] = red[0];
}

// ---- 6) deterministic final reduction -------------------------------------
__global__ __launch_bounds__(256) void k_final(const float* __restrict__ partial,
                                               float* __restrict__ out) {
  __shared__ float red[256];
  float s = 0.f;
  for (int t = threadIdx.x; t < 1024; t += 256) s += partial[t];
  red[threadIdx.x] = s;
  __syncthreads();
  for (int st = 128; st > 0; st >>= 1) {
    if (threadIdx.x < st) red[threadIdx.x] += red[threadIdx.x + st];
    __syncthreads();
  }
  if (threadIdx.x == 0) out[0] = red[0] * (1.0f / 262144.0f);
}

// ---------------------------------------------------------------------------
extern "C" void kernel_launch(void* const* d_in, const int* in_sizes, int n_in,
                              void* d_out, int out_size, void* d_ws,
                              size_t ws_size, hipStream_t stream) {
  (void)in_sizes; (void)n_in; (void)out_size; (void)ws_size;
  const float* x       = (const float*)d_in[0];   // (512,256,16,16) fp32
  const int*   ranking = (const int*)d_in[1];     // (512,16) int32
  const int*   choice  = (const int*)d_in[2];     // (512,) int32

  char* ws = (char*)d_ws;
  __bf16* xh    = (__bf16*)(ws);                        // 64 MB
  __bf16* xl    = (__bf16*)(ws + 67108864ull);          // 64 MB
  float*  gpart = (float*)(ws + 134217728ull);          // 16 MB
  int*    pos   = (int*)(ws + 150994944ull);            // 32 KB
  float*  sq    = (float*)(ws + 151027712ull);          // 2 KB
  float*  part  = (float*)(ws + 151029760ull);          // 4 KB

  k_convert<<<16384, 256, 0, stream>>>(x, xh, xl);      // 512*65536/(256*8)
  k_pos<<<2, 256, 0, stream>>>(ranking, pos);
  k_gram<<<16 * SPLITK, 256, 0, stream>>>(xh, xl, gpart);
  k_diag<<<2, 256, 0, stream>>>(gpart, sq);
  k_loss<<<1024, 256, 0, stream>>>(gpart, sq, pos, ranking, choice, part);
  k_final<<<1, 256, 0, stream>>>(part, (float*)d_out);
}
